// VisionTransformer_63625645523160
// MI455X (gfx1250) — matrix operand order repaired and checked by reading, not compiled, not run on hardware
//
#include <hip/hip_runtime.h>
#include <hip/hip_bf16.h>
#include <math.h>

// Problem constants (from reference)
#define Bc   1024
#define Nn   31
#define Dm   576
#define Hh   16
#define Ll   12
#define DHh  36
#define FFf  2304
#define NCc  1000
#define NEGV (-987654321.0f)
#define EPSV 1e-5f

typedef __bf16 bf16;
typedef __attribute__((ext_vector_type(16))) __bf16 v16bf;
typedef __attribute__((ext_vector_type(8)))  __bf16 v8bf;
typedef __attribute__((ext_vector_type(8)))  float  v8f;

// ---------------------------------------------------------------------------
// fp32 -> bf16 conversion (weights, once per launch)
// ---------------------------------------------------------------------------
__global__ __launch_bounds__(256) void f2bf_kernel(const float* __restrict__ in,
                                                   bf16* __restrict__ out, size_t n) {
  size_t i = (size_t)blockIdx.x * blockDim.x + threadIdx.x;
  size_t stride = (size_t)gridDim.x * blockDim.x;
  for (; i < n; i += stride) out[i] = (bf16)in[i];
}

// ---------------------------------------------------------------------------
// LayerNorm over D=576, one row per block; bf16 output (feeds WMMA GEMMs).
// inRowStride lets the final LN read only cls rows (stride N*D).
// ---------------------------------------------------------------------------
__global__ __launch_bounds__(256) void ln_kernel(const float* __restrict__ in, long inRowStride,
                                                 const float* __restrict__ g,
                                                 const float* __restrict__ be,
                                                 bf16* __restrict__ out) {
  __shared__ float row[Dm];
  __shared__ float red[256];
  const int r = blockIdx.x;
  const int t = threadIdx.x;
  const float* ip = in + (size_t)r * inRowStride;

  float s = 0.f;
  for (int i = t; i < Dm; i += 256) { float v = ip[i]; row[i] = v; s += v; }
  red[t] = s; __syncthreads();
  for (int o = 128; o > 0; o >>= 1) { if (t < o) red[t] += red[t + o]; __syncthreads(); }
  const float mean = red[0] * (1.0f / Dm);
  __syncthreads();

  float s2 = 0.f;
  for (int i = t; i < Dm; i += 256) { float d = row[i] - mean; s2 += d * d; }
  red[t] = s2; __syncthreads();
  for (int o = 128; o > 0; o >>= 1) { if (t < o) red[t] += red[t + o]; __syncthreads(); }
  const float inv = rsqrtf(red[0] * (1.0f / Dm) + EPSV);

  bf16* op = out + (size_t)r * Dm;
  for (int i = t; i < Dm; i += 256)
    op[i] = (bf16)((row[i] - mean) * inv * g[i] + be[i]);
}

// ---------------------------------------------------------------------------
// NT-GEMM with WMMA bf16->f32:  C[M x Nw] = act(A[MxK] * W[NwxK]^T + bias (+resid))
//
// Block: 256 threads = 8 waves as 2 (M) x 4 (N); block tile 64(M) x 128(N).
// Each wave owns a 32x32 C tile = 2x2 WMMA subtiles -> 4 v8f accumulators,
// giving 4 v_wmma per K-step against 2 LDS A-operand reads + 2 global B reads.
//
// A tile (64x32 bf16) is staged in LDS once per K-step (one b128 per thread),
// software-pipelined: the next K-slice is preloaded into registers before the
// barrier so the global latency hides behind the WMMAs.
//
// B operands come straight from global as contiguous 32B chunks per lane
// (K-major weight rows, contiguous k order as for A; lane half picks k-chunk
// 16*half..+15). Edge columns are handled by clamping the row index (loads
// always legal) and guarding only the epilogue stores -> branch-free loop.
// ---------------------------------------------------------------------------
__global__ __launch_bounds__(256) void gemm_bf16_wmma(
    const bf16* __restrict__ A, const bf16* __restrict__ W,
    const float* __restrict__ bias, const float* __restrict__ resid,
    float* __restrict__ outF, bf16* __restrict__ outB,
    int M, int Nw, int K, int act) {
  __shared__ bf16 As[64][32];

  const int t    = threadIdx.x;
  const int wave = t >> 5;
  const int lane = t & 31;
  const int half = lane >> 4;   // k-chunk selector per ISA operand layouts
  const int nloc = lane & 15;   // col within subtile (B/C), row within subtile (A)
  const int mw   = wave & 1;    // wave M position (0..1)
  const int nw   = wave >> 1;   // wave N position (0..3)

  const int m0 = blockIdx.y * 64;
  const int n0 = blockIdx.x * 128 + nw * 32;

  // B streams: two column subtiles per wave, clamped so loads are always legal.
  const int col0 = n0 + nloc;
  const int col1 = n0 + 16 + nloc;
  const int cc0 = col0 < Nw ? col0 : Nw - 1;
  const int cc1 = col1 < Nw ? col1 : Nw - 1;
  const bf16* wrow0 = W + (size_t)cc0 * K + half * 16;
  const bf16* wrow1 = W + (size_t)cc1 * K + half * 16;

  // Cooperative A staging: thread t owns row t/4, 8-elem chunk (t%4)*8.
  const int arow  = t >> 2;
  const int acol8 = (t & 3) * 8;
  const bf16* aptr = A + (size_t)(m0 + arow) * K + acol8;

  // A-operand LDS read pointers for the two M subtiles of this wave.
  const bf16* arp0 = &As[mw * 32 + nloc][0];
  const bf16* arp1 = &As[mw * 32 + 16 + nloc][0];

  v8f c00 = {}, c01 = {}, c10 = {}, c11 = {};

  v8bf astage = *(const v8bf*)(aptr);   // preload K-slice 0
  for (int k0 = 0; k0 < K; k0 += 32) {
    __syncthreads();
    *(v8bf*)(&As[arow][acol8]) = astage;
    __syncthreads();
    if (k0 + 32 < K) astage = *(const v8bf*)(aptr + k0 + 32);  // pipeline next slice

    // A operands (contiguous k order, matching B): elems 0..15 -> k = 16*half+0..15,
    // row = nloc within subtile.
    v8bf a0lo = *(const v8bf*)(arp0 + 16 * half);
    v8bf a0hi = *(const v8bf*)(arp0 + 16 * half + 8);
    v8bf a1lo = *(const v8bf*)(arp1 + 16 * half);
    v8bf a1hi = *(const v8bf*)(arp1 + 16 * half + 8);
    v16bf a0 = __builtin_shufflevector(a0lo, a0hi, 0, 1, 2, 3, 4, 5, 6, 7,
                                                   8, 9, 10, 11, 12, 13, 14, 15);
    v16bf a1 = __builtin_shufflevector(a1lo, a1hi, 0, 1, 2, 3, 4, 5, 6, 7,
                                                   8, 9, 10, 11, 12, 13, 14, 15);

    v16bf b0 = *(const v16bf*)(wrow0 + k0);
    v16bf b1 = *(const v16bf*)(wrow1 + k0);
    __builtin_prefetch(wrow0 + k0 + 256, 0, 0);  // global_prefetch_b8 weight stream
    __builtin_prefetch(wrow1 + k0 + 256, 0, 0);

    c00 = __builtin_amdgcn_wmma_f32_16x16x32_bf16(false, a0, false, b0, (short)0, c00, false, false);
    c01 = __builtin_amdgcn_wmma_f32_16x16x32_bf16(false, a0, false, b1, (short)0, c01, false, false);
    c10 = __builtin_amdgcn_wmma_f32_16x16x32_bf16(false, a1, false, b0, (short)0, c10, false, false);
    c11 = __builtin_amdgcn_wmma_f32_16x16x32_bf16(false, a1, false, b1, (short)0, c11, false, false);
  }

  // C layout: vgpr j -> row subtile_base + 8*half + j, col = subtile_base + nloc
  const bool ok0 = (col0 < Nw);
  const bool ok1 = (col1 < Nw);
  const float bia0 = (bias && ok0) ? bias[col0] : 0.f;
  const float bia1 = (bias && ok1) ? bias[col1] : 0.f;
#pragma unroll
  for (int mi = 0; mi < 2; mi++) {
    v8f cm0 = mi ? c10 : c00;
    v8f cm1 = mi ? c11 : c01;
#pragma unroll
    for (int j = 0; j < 8; j++) {
      const int row = m0 + mw * 32 + mi * 16 + half * 8 + j;
#pragma unroll
      for (int ni = 0; ni < 2; ni++) {
        const bool ok = ni ? ok1 : ok0;
        if (!ok) continue;
        const int col = ni ? col1 : col0;
        float v = ni ? cm1[j] : cm0[j];
        v += ni ? bia1 : bia0;
        const size_t idx = (size_t)row * Nw + col;
        if (resid) v += resid[idx];
        if (act) v = 0.5f * v * (1.0f + erff(v * 0.70710678118654752f));  // exact GELU
        if (outF) outF[idx] = v;
        if (outB) outB[idx] = (bf16)v;
      }
    }
  }
}

// ---------------------------------------------------------------------------
// Attention: one block per (batch, head). N=31, DH=36 — negligible FLOPs, so
// a VALU kernel: LDS-staged q/k/v, masked-diag softmax in f32, bf16 output.
// ---------------------------------------------------------------------------
__global__ __launch_bounds__(128) void attn_kernel(const bf16* __restrict__ qkv,
                                                   const float* __restrict__ scale_all,
                                                   int layer, bf16* __restrict__ o) {
  __shared__ float qs[Nn][DHh + 1];
  __shared__ float ks[Nn][DHh + 1];
  __shared__ float vs[Nn][DHh + 1];
  __shared__ float sc[Nn][Nn + 1];

  const int bh = blockIdx.x;
  const int b  = bh / Hh;
  const int hd = bh % Hh;
  const int t  = threadIdx.x;

  for (int idx = t; idx < Nn * DHh; idx += 128) {
    const int i = idx / DHh, d = idx % DHh;
    const size_t base = (size_t)(b * Nn + i) * (3 * Dm) + hd * DHh + d;
    qs[i][d] = (float)qkv[base];
    ks[i][d] = (float)qkv[base + Dm];
    vs[i][d] = (float)qkv[base + 2 * Dm];
  }
  __syncthreads();

  const float scl = scale_all[layer * Hh + hd];
  for (int idx = t; idx < Nn * Nn; idx += 128) {
    const int i = idx / Nn, j = idx % Nn;
    float s = 0.f;
#pragma unroll 4
    for (int d = 0; d < DHh; d++) s += qs[i][d] * ks[j][d];
    sc[i][j] = (i == j) ? NEGV : s * scl;
  }
  __syncthreads();

  if (t < Nn) {
    float mx = -3.4e38f;
    for (int j = 0; j < Nn; j++) mx = fmaxf(mx, sc[t][j]);
    float sum = 0.f;
    for (int j = 0; j < Nn; j++) { float p = expf(sc[t][j] - mx); sc[t][j] = p; sum += p; }
    const float inv = 1.0f / sum;
    for (int j = 0; j < Nn; j++) sc[t][j] *= inv;
  }
  __syncthreads();

  for (int idx = t; idx < Nn * DHh; idx += 128) {
    const int i = idx / DHh, d = idx % DHh;
    float acc = 0.f;
    for (int j = 0; j < Nn; j++) acc += sc[i][j] * vs[j][d];
    o[(size_t)(b * Nn + i) * Dm + hd * DHh + d] = (bf16)acc;
  }
}

// ---------------------------------------------------------------------------
// Host orchestration
// ---------------------------------------------------------------------------
extern "C" void kernel_launch(void* const* d_in, const int* in_sizes, int n_in,
                              void* d_out, int out_size, void* d_ws, size_t ws_size,
                              hipStream_t stream) {
  const float* x      = (const float*)d_in[0];
  const float* qkv_w  = (const float*)d_in[1];
  const float* scale  = (const float*)d_in[2];
  const float* out_w  = (const float*)d_in[3];
  const float* out_b  = (const float*)d_in[4];
  const float* ln1_w  = (const float*)d_in[5];
  const float* ln1_b  = (const float*)d_in[6];
  const float* ln2_w  = (const float*)d_in[7];
  const float* ln2_b  = (const float*)d_in[8];
  const float* fc1_w  = (const float*)d_in[9];
  const float* fc1_b  = (const float*)d_in[10];
  const float* fc2_w  = (const float*)d_in[11];
  const float* fc2_b  = (const float*)d_in[12];
  const float* norm_w = (const float*)d_in[13];
  const float* norm_b = (const float*)d_in[14];
  const float* head_w = (const float*)d_in[15];
  const float* head_b = (const float*)d_in[16];

  const int M    = Bc * Nn;   // 31744 token rows
  const int My64 = M / 64;    // 496 row tiles

  // Workspace carve (256B aligned)
  char* base = (char*)d_ws;
  size_t off = 0;
  auto carve = [&](size_t bytes) -> char* {
    char* p = base + off;
    off = (off + bytes + 255) & ~(size_t)255;
    return p;
  };
  bf16* w_qkv  = (bf16*)carve((size_t)Ll * 3 * Dm * Dm * sizeof(bf16));
  bf16* w_out  = (bf16*)carve((size_t)Ll * Dm * Dm * sizeof(bf16));
  bf16* w_fc1  = (bf16*)carve((size_t)Ll * FFf * Dm * sizeof(bf16));
  bf16* w_fc2  = (bf16*)carve((size_t)Ll * Dm * FFf * sizeof(bf16));
  bf16* w_head = (bf16*)carve((size_t)NCc * Dm * sizeof(bf16));
  float* h     = (float*)carve((size_t)M * Dm * sizeof(float));
  bf16* ybuf   = (bf16*)carve((size_t)M * Dm * sizeof(bf16));   // y / attn-out / cls
  bf16* qkvb   = (bf16*)carve((size_t)M * 3 * Dm * sizeof(bf16));
  bf16* mbuf   = (bf16*)carve((size_t)M * FFf * sizeof(bf16));
  (void)ws_size; (void)in_sizes; (void)n_in; (void)out_size;

  // Weight conversion fp32 -> bf16
  auto conv = [&](const float* src, bf16* dst, size_t n) {
    int blocks = (int)((n + 255) / 256);
    f2bf_kernel<<<blocks, 256, 0, stream>>>(src, dst, n);
  };
  conv(qkv_w,  w_qkv,  (size_t)Ll * 3 * Dm * Dm);
  conv(out_w,  w_out,  (size_t)Ll * Dm * Dm);
  conv(fc1_w,  w_fc1,  (size_t)Ll * FFf * Dm);
  conv(fc2_w,  w_fc2,  (size_t)Ll * Dm * FFf);
  conv(head_w, w_head, (size_t)NCc * Dm);

  // h = x
  hipMemcpyAsync(h, x, (size_t)M * Dm * sizeof(float), hipMemcpyDeviceToDevice, stream);

  auto gxy = [](int Nw) { return (Nw + 127) / 128; };

  for (int l = 0; l < Ll; l++) {
    // y = LN1(h)
    ln_kernel<<<M, 256, 0, stream>>>(h, Dm, ln1_w + (size_t)l * Dm, ln1_b + (size_t)l * Dm, ybuf);
    // qkv = y @ Wqkv^T   (no bias)
    gemm_bf16_wmma<<<dim3(gxy(3 * Dm), My64), 256, 0, stream>>>(
        ybuf, w_qkv + (size_t)l * 3 * Dm * Dm, nullptr, nullptr,
        nullptr, qkvb, M, 3 * Dm, Dm, 0);
    // o = softmax(q k^T * s, diag->NEG) v
    attn_kernel<<<Bc * Hh, 128, 0, stream>>>(qkvb, scale, l, ybuf);
    // h = h + o @ Wout^T + bout
    gemm_bf16_wmma<<<dim3(gxy(Dm), My64), 256, 0, stream>>>(
        ybuf, w_out + (size_t)l * Dm * Dm, out_b + (size_t)l * Dm, h,
        h, nullptr, M, Dm, Dm, 0);
    // y2 = LN2(h)
    ln_kernel<<<M, 256, 0, stream>>>(h, Dm, ln2_w + (size_t)l * Dm, ln2_b + (size_t)l * Dm, ybuf);
    // m = gelu(y2 @ W1^T + b1)
    gemm_bf16_wmma<<<dim3(gxy(FFf), My64), 256, 0, stream>>>(
        ybuf, w_fc1 + (size_t)l * FFf * Dm, fc1_b + (size_t)l * FFf, nullptr,
        nullptr, mbuf, M, FFf, Dm, 1);
    // h = h + m @ W2^T + b2
    gemm_bf16_wmma<<<dim3(gxy(Dm), My64), 256, 0, stream>>>(
        mbuf, w_fc2 + (size_t)l * Dm * FFf, fc2_b + (size_t)l * Dm, h,
        h, nullptr, M, Dm, FFf, 0);
  }

  // Final LN on cls rows only (row stride N*D), then head GEMM -> d_out (f32)
  ln_kernel<<<Bc, 256, 0, stream>>>(h, (long)Nn * Dm, norm_w, norm_b, ybuf);
  gemm_bf16_wmma<<<dim3(gxy(NCc), Bc / 64), 256, 0, stream>>>(
      ybuf, w_head, head_b, nullptr, (float*)d_out, nullptr, Bc, NCc, Dm, 0);
}
